// MOELayer_19112604467571
// MI455X (gfx1250) — compile-verified, matrix operand order checked by reading, not run on hardware
//
#include <hip/hip_runtime.h>
#include <math.h>

// ---- problem constants (match reference) ----
#define TOK 4096   // B*S
#define DM  1024   // model dim D
#define NE  8      // experts
#define TOPK 2
#define HID 4096   // expert hidden H
#define CAP 1024   // int(K * 1.0 * ceil(T/E)) = 2*512

typedef __bf16 v16bf __attribute__((ext_vector_type(16)));
typedef float  v8f   __attribute__((ext_vector_type(8)));

union Frag32 { v16bf v; uint4 q[2]; };

__device__ __forceinline__ unsigned short f2bf(float f) {
  unsigned u = __float_as_uint(f);
  u += 0x7fffu + ((u >> 16) & 1u);   // round-to-nearest-even
  return (unsigned short)(u >> 16);
}

// ---------------- gating: logits = x @ wg, softmax, top2, renorm ----------------
__global__ __launch_bounds__(256) void gate_kernel(
    const float* __restrict__ x, const float* __restrict__ wg,
    float* __restrict__ gates, int* __restrict__ idxk)
{
  int token = (blockIdx.x * 256 + threadIdx.x) >> 5;   // one wave32 per token
  int lane  = threadIdx.x & 31;
  const float* xt = x + (size_t)token * DM;
  float acc[NE];
  #pragma unroll
  for (int e = 0; e < NE; ++e) acc[e] = 0.f;
  for (int d = lane; d < DM; d += 32) {
    float xv = xt[d];
    const float* wr = wg + (size_t)d * NE;
    float4 w0 = *(const float4*)(wr);
    float4 w1 = *(const float4*)(wr + 4);
    acc[0] += xv * w0.x; acc[1] += xv * w0.y; acc[2] += xv * w0.z; acc[3] += xv * w0.w;
    acc[4] += xv * w1.x; acc[5] += xv * w1.y; acc[6] += xv * w1.z; acc[7] += xv * w1.w;
  }
  #pragma unroll
  for (int off = 16; off > 0; off >>= 1)
    #pragma unroll
    for (int e = 0; e < NE; ++e)
      acc[e] += __shfl_xor(acc[e], off, 32);
  if (lane == 0) {
    float m = acc[0];
    #pragma unroll
    for (int e = 1; e < NE; ++e) m = fmaxf(m, acc[e]);
    float p[NE], s = 0.f;
    #pragma unroll
    for (int e = 0; e < NE; ++e) { p[e] = __expf(acc[e] - m); s += p[e]; }
    float inv = 1.f / s;
    #pragma unroll
    for (int e = 0; e < NE; ++e) p[e] *= inv;
    int i0 = 0; float b0 = p[0];
    #pragma unroll
    for (int e = 1; e < NE; ++e) if (p[e] > b0) { b0 = p[e]; i0 = e; }
    int i1 = -1; float b1 = -1.f;
    #pragma unroll
    for (int e = 0; e < NE; ++e) if (e != i0 && p[e] > b1) { b1 = p[e]; i1 = e; }
    float rs = 1.f / (b0 + b1);
    gates[token * TOPK + 0] = b0 * rs;
    gates[token * TOPK + 1] = b1 * rs;
    idxk[token * TOPK + 0] = i0;
    idxk[token * TOPK + 1] = i1;
  }
}

// -------- GShard slot-major location assignment (exact sequential cumsum) --------
__global__ void assign_kernel(const int* __restrict__ idxk, int* __restrict__ loc)
{
  int lane = threadIdx.x;                 // single wave32
  int cnt[NE];
  #pragma unroll
  for (int e = 0; e < NE; ++e) cnt[e] = 0;
  for (int s0 = 0; s0 < TOPK * TOK; s0 += 32) {
    int s = s0 + lane;
    int k = s / TOK, t = s % TOK;
    int e = idxk[t * TOPK + k];
    unsigned before = (1u << lane) - 1u;
    int myloc = 0;
    #pragma unroll
    for (int ee = 0; ee < NE; ++ee) {
      unsigned msk = (unsigned)(__ballot(e == ee) & 0xffffffffull);
      if (ee == e) myloc = cnt[ee] + __popc(msk & before);
      cnt[ee] += __popc(msk);
    }
    loc[t * TOPK + k] = myloc;
  }
}

// ---------------- zero fill (graph-capture-safe memset) ----------------
__global__ void zero_u4(uint4* __restrict__ p, size_t n4)
{
  size_t i = (size_t)blockIdx.x * 256 + threadIdx.x;
  if (i < n4) p[i] = make_uint4(0u, 0u, 0u, 0u);
}

// ---------------- dispatch: scatter x (fp32 -> bf16) into [E,cap,D] ----------------
__global__ __launch_bounds__(256) void dispatch_kernel(
    const float* __restrict__ x, const int* __restrict__ idxk,
    const int* __restrict__ loc, unsigned short* __restrict__ disp)
{
  int t = blockIdx.x >> 1, k = blockIdx.x & 1;
  int l = loc[t * TOPK + k];
  if (l >= CAP) return;                      // capacity-dropped slot
  int e = idxk[t * TOPK + k];
  size_t dst = (((size_t)e * CAP + l) << 10);          // *DM
  float4 v = ((const float4*)(x + ((size_t)t << 10)))[threadIdx.x];
  uint2 pk;
  pk.x = (unsigned)f2bf(v.x) | ((unsigned)f2bf(v.y) << 16);
  pk.y = (unsigned)f2bf(v.z) | ((unsigned)f2bf(v.w) << 16);
  *(uint2*)(disp + dst + (size_t)threadIdx.x * 4) = pk;
}

// ------- weight prep: fp32 [E][K][N] -> bf16 transposed [E][N][K] (one cvt/elem) -------
// 64x64 tiles through LDS; coalesced b128 reads and writes.
__global__ __launch_bounds__(256) void transpose_cvt_kernel(
    const float* __restrict__ W, unsigned short* __restrict__ WT, int K, int N)
{
  __shared__ __align__(16) unsigned short Ls[64 * 72];   // pad stride 72 (144B, 16B-aligned)
  const int e  = blockIdx.z;
  const int n0 = blockIdx.x * 64;
  const int k0 = blockIdx.y * 64;
  const float* We = W + (size_t)e * K * N;
  unsigned short* WTe = WT + (size_t)e * K * N;
  // read 64(k) x 64(n) fp32 coalesced, convert, scatter into Ls[n][k]
  #pragma unroll
  for (int i = 0; i < 4; ++i) {
    int u  = threadIdx.x + i * 256;
    int kk = u >> 4;            // 0..63
    int nl = (u & 15) * 4;      // 0..60
    float4 w = *(const float4*)(We + (size_t)(k0 + kk) * N + n0 + nl);
    Ls[(nl + 0) * 72 + kk] = f2bf(w.x);
    Ls[(nl + 1) * 72 + kk] = f2bf(w.y);
    Ls[(nl + 2) * 72 + kk] = f2bf(w.z);
    Ls[(nl + 3) * 72 + kk] = f2bf(w.w);
  }
  __syncthreads();
  // write 64(n) x 64(k) bf16 coalesced (uint4 = 8 elems)
  #pragma unroll
  for (int i = 0; i < 2; ++i) {
    int u  = threadIdx.x + i * 256;
    int nn = u >> 3;            // 0..63
    int c  = (u & 7) * 8;       // 0..56
    uint4 q = *(const uint4*)(&Ls[nn * 72 + c]);
    *(uint4*)(WTe + (size_t)(n0 + nn) * K + k0 + c) = q;
  }
}

// ---- async copy helpers: gfx1250 GLOBAL_LOAD_ASYNC_TO_LDS_B128 (ASYNCcnt path) ----
__device__ __forceinline__ void async_copy_b128(const unsigned short* gptr, unsigned lds_off)
{
  unsigned long long ga = (unsigned long long)(uintptr_t)gptr;
  asm volatile("global_load_async_to_lds_b128 %0, %1, off"
               :: "v"(lds_off), "v"(ga) : "memory");
}
__device__ __forceinline__ void wait_async0()
{
  asm volatile("s_wait_asynccnt 0x0" ::: "memory");
}

// ---------------- bf16 WMMA GEMM: C[e] = act(A[e][M,K] * BT[e][N,K]^T + bias[e]) ----------------
// A bf16 row-major [M,K]; BT bf16 [N,K] (pre-transposed weights) -> both tiles are pure
// 16B copies into LDS, moved with async global->LDS, double-buffered.
// Block tile 128x128x32, 8 waves, each wave 32x64 = 2x4 WMMA 16x16 tiles.
template<int RELU_BF16>
__global__ __launch_bounds__(256) void moe_gemm_bf16(
    const unsigned short* __restrict__ A, const unsigned short* __restrict__ BT,
    const float* __restrict__ bias, void* __restrict__ Cout,
    int M, int N, int K)
{
  __shared__ __align__(16) unsigned short As[2][128 * 32];  // [row][k]
  __shared__ __align__(16) unsigned short Bs[2][128 * 32];  // [col][k]
  const int e  = blockIdx.z;
  const int m0 = blockIdx.y * 128;
  const int n0 = blockIdx.x * 128;
  const unsigned short* Ae  = A  + (size_t)e * M * K;
  const unsigned short* BTe = BT + (size_t)e * N * K;
  const int tid  = threadIdx.x;
  const int wave = tid >> 5, lane = tid & 31;
  const int wm = (wave & 3) * 32;        // 4 waves along M
  const int wn = (wave >> 2) * 64;       // 2 waves along N
  const int grp = lane >> 4;             // half-wave group (K-chunk select)
  const int lc  = lane & 15;

  // per-thread staging units: unit u in [0,512): row=u>>2 (0..127), ch=(u&3)*8
  const int u0   = tid * 2;
  const int row0 = u0 >> 2,       ch0 = (u0 & 3) * 8;
  const int row1 = (u0 + 1) >> 2, ch1 = ((u0 + 1) & 3) * 8;

  v8f acc[2][4];
  #pragma unroll
  for (int mt = 0; mt < 2; ++mt)
    #pragma unroll
    for (int nt = 0; nt < 4; ++nt)
      acc[mt][nt] = (v8f){0.f,0.f,0.f,0.f,0.f,0.f,0.f,0.f};

  // issue async copies of the 128x32 A and B tiles for K-offset kt into buffer `buf`
  auto issue_stage = [&](int buf, int kt) {
    async_copy_b128(Ae  + (size_t)(m0 + row0) * K + kt + ch0,
                    (unsigned)(uintptr_t)&As[buf][row0 * 32 + ch0]);
    async_copy_b128(Ae  + (size_t)(m0 + row1) * K + kt + ch1,
                    (unsigned)(uintptr_t)&As[buf][row1 * 32 + ch1]);
    async_copy_b128(BTe + (size_t)(n0 + row0) * K + kt + ch0,
                    (unsigned)(uintptr_t)&Bs[buf][row0 * 32 + ch0]);
    async_copy_b128(BTe + (size_t)(n0 + row1) * K + kt + ch1,
                    (unsigned)(uintptr_t)&Bs[buf][row1 * 32 + ch1]);
  };

  issue_stage(0, 0);
  int buf = 0;
  for (int kt = 0; kt < K; kt += 32) {
    wait_async0();        // my async writes (this buffer) done
    __syncthreads();      // everyone's writes visible
    if (kt + 32 < K) issue_stage(buf ^ 1, kt + 32);   // overlap next tile with WMMAs

    // A fragment: lanes 0-15 rows, K chunks {grp*8..+7} and {16+grp*8..+7}
    Frag32 a[2];
    #pragma unroll
    for (int mt = 0; mt < 2; ++mt) {
      int row = wm + mt * 16 + lc;
      a[mt].q[0] = *(const uint4*)(&As[buf][row * 32 + grp * 8]);
      a[mt].q[1] = *(const uint4*)(&As[buf][row * 32 + 16 + grp * 8]);
    }
    #pragma unroll
    for (int nt = 0; nt < 4; ++nt) {
      Frag32 b;
      int col = wn + nt * 16 + lc;
      b.q[0] = *(const uint4*)(&Bs[buf][col * 32 + grp * 8]);
      b.q[1] = *(const uint4*)(&Bs[buf][col * 32 + 16 + grp * 8]);
      #pragma unroll
      for (int mt = 0; mt < 2; ++mt)
        acc[mt][nt] = __builtin_amdgcn_wmma_f32_16x16x32_bf16(
            false, a[mt].v, false, b.v, (short)0, acc[mt][nt], false, false);
    }
    __syncthreads();      // done reading this buffer before it is overwritten
    buf ^= 1;
  }

  // epilogue: C frag layout f32 16x16 -> row = grp*8 + v, col = lc
  const float* be = bias + (size_t)e * N;
  #pragma unroll
  for (int mt = 0; mt < 2; ++mt) {
    #pragma unroll
    for (int nt = 0; nt < 4; ++nt) {
      int gcol = n0 + wn + nt * 16 + lc;
      float bv = be[gcol];
      #pragma unroll
      for (int v = 0; v < 8; ++v) {
        int grow = m0 + wm + mt * 16 + grp * 8 + v;
        float val = acc[mt][nt][v] + bv;
        size_t o = (size_t)e * M * N + (size_t)grow * N + gcol;
        if (RELU_BF16) ((unsigned short*)Cout)[o] = f2bf(fmaxf(val, 0.f));
        else           ((float*)Cout)[o] = val;
      }
    }
  }
}

// ---------------- combine: y[t] = sum_k gate*valid * out[idx,loc] ----------------
__global__ __launch_bounds__(256) void combine_kernel(
    const float* __restrict__ eo, const float* __restrict__ gates,
    const int* __restrict__ idxk, const int* __restrict__ loc,
    float* __restrict__ y)
{
  int t = blockIdx.x;
  float g0 = gates[t * TOPK + 0], g1 = gates[t * TOPK + 1];
  int e0 = idxk[t * TOPK + 0], e1 = idxk[t * TOPK + 1];
  int l0 = loc[t * TOPK + 0],  l1 = loc[t * TOPK + 1];
  float v0 = (l0 < CAP) ? g0 : 0.f;
  float v1 = (l1 < CAP) ? g1 : 0.f;
  l0 = (l0 < CAP) ? l0 : (CAP - 1);
  l1 = (l1 < CAP) ? l1 : (CAP - 1);
  const float4* p0 = (const float4*)(eo + (((size_t)e0 * CAP + l0) << 10));
  const float4* p1 = (const float4*)(eo + (((size_t)e1 * CAP + l1) << 10));
  int i = threadIdx.x;
  float4 a = p0[i], b = p1[i];
  float4 r;
  r.x = v0 * a.x + v1 * b.x;
  r.y = v0 * a.y + v1 * b.y;
  r.z = v0 * a.z + v1 * b.z;
  r.w = v0 * a.w + v1 * b.w;
  ((float4*)(y + ((size_t)t << 10)))[i] = r;
}

extern "C" void kernel_launch(void* const* d_in, const int* in_sizes, int n_in,
                              void* d_out, int out_size, void* d_ws, size_t ws_size,
                              hipStream_t stream)
{
  const float* x  = (const float*)d_in[0];
  const float* wg = (const float*)d_in[1];
  const float* w1 = (const float*)d_in[2];
  const float* b1 = (const float*)d_in[3];
  const float* w2 = (const float*)d_in[4];
  const float* b2 = (const float*)d_in[5];
  float* y = (float*)d_out;

  char* ws = (char*)d_ws;
  size_t off = 0;
  auto take = [&](size_t bytes) {
    char* p = ws + off;
    off = (off + bytes + 255) & ~(size_t)255;
    return p;
  };
  float*          gates = (float*)take((size_t)TOK * TOPK * sizeof(float));
  int*            idxk  = (int*)take((size_t)TOK * TOPK * sizeof(int));
  int*            loc   = (int*)take((size_t)TOK * TOPK * sizeof(int));
  unsigned short* disp  = (unsigned short*)take((size_t)NE * CAP * DM * 2);   // 16 MB bf16
  unsigned short* hbuf  = (unsigned short*)take((size_t)NE * CAP * HID * 2);  // 64 MB bf16
  float*          eo    = (float*)take((size_t)NE * CAP * DM * 4);            // 32 MB f32
  unsigned short* w1T   = (unsigned short*)take((size_t)NE * DM * HID * 2);   // 64 MB bf16 [E][H][D]
  unsigned short* w2T   = (unsigned short*)take((size_t)NE * HID * DM * 2);   // 64 MB bf16 [E][D][H]

  gate_kernel<<<TOK / 8, 256, 0, stream>>>(x, wg, gates, idxk);
  assign_kernel<<<1, 32, 0, stream>>>(idxk, loc);

  size_t n4 = ((size_t)NE * CAP * DM * 2) / 16;
  zero_u4<<<(unsigned)((n4 + 255) / 256), 256, 0, stream>>>((uint4*)disp, n4);
  dispatch_kernel<<<TOK * TOPK, 256, 0, stream>>>(x, idxk, loc, disp);

  // one-time weight downconvert + transpose: fp32 [K,N] -> bf16 [N,K]
  transpose_cvt_kernel<<<dim3(HID / 64, DM / 64, NE), 256, 0, stream>>>(w1, w1T, DM, HID);
  transpose_cvt_kernel<<<dim3(DM / 64, HID / 64, NE), 256, 0, stream>>>(w2, w2T, HID, DM);

  // h = relu(disp @ w1 + b1) : per-expert M=CAP, K=DM, N=HID, stored bf16
  moe_gemm_bf16<1><<<dim3(HID / 128, CAP / 128, NE), 256, 0, stream>>>(
      disp, w1T, b1, (void*)hbuf, CAP, HID, DM);
  // out = h @ w2 + b2 : per-expert M=CAP, K=HID, N=DM, stored f32
  moe_gemm_bf16<0><<<dim3(DM / 128, CAP / 128, NE), 256, 0, stream>>>(
      hbuf, w2T, b2, (void*)eo, CAP, DM, HID);

  combine_kernel<<<TOK, 256, 0, stream>>>(eo, gates, idxk, loc, y);
}
